// DistogramLoss_4861902979516
// MI455X (gfx1250) — compile-verified, matrix operand order checked by reading, not run on hardware
//
#include <hip/hip_runtime.h>
#include <math.h>

// ---- shapes / constants from the reference ----
#define D_MODEL   512
#define D_LOW     64
#define NUM_BINS  39
#define KPAD      48            // 3 tiles of 16 bins (39 padded to 48)
#define DIST_MIN  2.0f
#define INV_BINW  1.9f          // (NUM_BINS-1)/(DIST_MAX-DIST_MIN) = 38/20
#define LN_EPS    1e-5f
#define NEG_PAD   (-2e30f)      // bias for padded bins -> logit ~ -inf

typedef __attribute__((ext_vector_type(16))) __bf16 v16bf;
typedef __attribute__((ext_vector_type(8)))  float  v8f;

// f32 -> bf16 raw bits, round-to-nearest-even
__device__ __forceinline__ unsigned short f2bf(float f) {
    union { float f; unsigned u; } v; v.f = f;
    unsigned r = v.u + 0x7fffu + ((v.u >> 16) & 1u);
    return (unsigned short)(r >> 16);
}
__device__ __forceinline__ __bf16 bfbits(unsigned short u) {
    union { unsigned short u; __bf16 b; } c; c.u = u; return c.b;
}

// Cross-half (lane ^ 16) exchange. v_permlanex16_b32 = pure VALU, no DS pipe.
#if __has_builtin(__builtin_amdgcn_permlanex16)
__device__ __forceinline__ float xhalf(float v) {
    int r = __builtin_amdgcn_permlanex16(__float_as_int(v), __float_as_int(v),
                                         0x76543210, 0xfedcba98, false, false);
    return __int_as_float(r);
}
#else
__device__ __forceinline__ float xhalf(float v) { return __shfl_xor(v, 16, 32); }
#endif

__device__ __forceinline__ float warp_red_sum32(float v) {
    #pragma unroll
    for (int m = 1; m < 32; m <<= 1) v += __shfl_xor(v, m, 32);
    return v;
}

// ---------------------------------------------------------------------------
// Kernel 0: zero the 4-float accumulator block (ce0,cnt0,ce1,cnt1)
// ---------------------------------------------------------------------------
__global__ void k_init(float* acc4) {
    if (threadIdx.x < 4) acc4[threadIdx.x] = 0.0f;
}

// ---------------------------------------------------------------------------
// Kernel 1: LayerNorm + two 512->64 projections. One block (256 thr) / token.
// U stored as bf16 (WMMA operand), V kept f32 (scaled by Wb later).
// ---------------------------------------------------------------------------
__global__ __launch_bounds__(256) void k_lnproj(
    const float* __restrict__ hres,
    const float* __restrict__ lnw, const float* __restrict__ lnb,
    const float* __restrict__ wuw, const float* __restrict__ wub,
    const float* __restrict__ wvw, const float* __restrict__ wvb,
    unsigned short* __restrict__ Ubf, float* __restrict__ Vf)
{
    __shared__ float sh[D_MODEL];
    __shared__ float red[8];
    const int tok = blockIdx.x;
    const int tid = threadIdx.x;
    const float* hp = hres + (size_t)tok * D_MODEL;

    float h0 = hp[tid], h1 = hp[tid + 256];

    // mean
    float s = warp_red_sum32(h0 + h1);
    if ((tid & 31) == 0) red[tid >> 5] = s;
    __syncthreads();
    if (tid < 8) {
        float v = red[tid];
        #pragma unroll
        for (int m = 1; m < 8; m <<= 1) v += __shfl_xor(v, m, 32);
        if (tid == 0) red[0] = v;
    }
    __syncthreads();
    const float mu = red[0] * (1.0f / D_MODEL);
    __syncthreads();

    // variance
    float d0 = h0 - mu, d1 = h1 - mu;
    float s2 = warp_red_sum32(d0 * d0 + d1 * d1);
    if ((tid & 31) == 0) red[tid >> 5] = s2;
    __syncthreads();
    if (tid < 8) {
        float v = red[tid];
        #pragma unroll
        for (int m = 1; m < 8; m <<= 1) v += __shfl_xor(v, m, 32);
        if (tid == 0) red[0] = v;
    }
    __syncthreads();
    const float inv = rsqrtf(red[0] * (1.0f / D_MODEL) + LN_EPS);

    sh[tid]       = d0 * inv * lnw[tid]       + lnb[tid];
    sh[tid + 256] = d1 * inv * lnw[tid + 256] + lnb[tid + 256];
    __syncthreads();

    if (tid < 128) {
        const int  o   = tid & 63;
        const bool isU = tid < 64;
        const float* W = (isU ? wuw : wvw) + (size_t)o * D_MODEL;
        float acc = isU ? wub[o] : wvb[o];
        #pragma unroll 8
        for (int c = 0; c < D_MODEL; ++c) acc += sh[c] * W[c];
        if (isU) Ubf[(size_t)tok * D_LOW + o] = f2bf(acc);
        else     Vf [(size_t)tok * D_LOW + o] = acc;
    }
}

// ---------------------------------------------------------------------------
// Kernel 2: main loss. Block = 256 thr (8 waves). Each wave owns one j.
// A-operand = T[j] (48 padded bins x 64, = V[j,c]*Wb[k,c], bf16) built once;
// B-operand = U i-tile (64 x 16 i's). C[k][i]: each lane owns one (i,j) pair
// with 24 of the 48 bins in registers -> softmax is register-space VALU plus
// 3 v_permlanex16 cross-half exchanges per pair. No DS ops in the hot loop.
// ---------------------------------------------------------------------------
__global__ __launch_bounds__(256) void k_loss(
    const unsigned short* __restrict__ Ubf, const float* __restrict__ Vf,
    const float* __restrict__ xtrue, const float* __restrict__ pmask,
    const float* __restrict__ wbw,   const float* __restrict__ wbb,
    float* __restrict__ acc4, int N, int jgPerB, int itilesPerSplit)
{
    __shared__ float sWb[NUM_BINS * D_LOW];
    __shared__ float sBias[KPAD];
    __shared__ float sV[8][D_LOW];
    __shared__ float sCe[256];
    __shared__ float sCnt[256];

    const int tid  = threadIdx.x;
    const int w    = tid >> 5;
    const int lane = tid & 31;
    const int h    = lane >> 4;     // lane half: selects K-subset / bin subset
    const int col  = lane & 15;     // A-row (bin) and B/C-column (i) index
    const int bb   = blockIdx.x / jgPerB;
    const int jg   = blockIdx.x % jgPerB;
    const int j    = jg * 8 + w;
    const int bN   = bb * N;

    for (int idx = tid; idx < NUM_BINS * D_LOW; idx += 256) sWb[idx] = wbw[idx];
    if (tid < KPAD) sBias[tid] = (tid < NUM_BINS) ? wbb[tid] : 0.0f;
    sV[w][lane]      = Vf[(size_t)(bN + j) * D_LOW + lane];
    sV[w][lane + 32] = Vf[(size_t)(bN + j) * D_LOW + lane + 32];
    __syncthreads();

    // ---- build A fragments once per wave: T[j] = V[j,c]*Wb[k,c], 48x64 ----
    v16bf Tf[3][2];
    #pragma unroll
    for (int t = 0; t < 3; ++t) {
        const int  kk = t * 16 + col;
        const bool ok = kk < NUM_BINS;
        #pragma unroll
        for (int cbi = 0; cbi < 2; ++cbi) {
            const int b1 = cbi * 32 + h * 8;
            const int b2 = b1 + 16;
            #pragma unroll
            for (int e = 0; e < 8; ++e) {
                float p1 = ok ? sV[w][b1 + e] * sWb[kk * D_LOW + b1 + e] : 0.0f;
                float p2 = ok ? sV[w][b2 + e] * sWb[kk * D_LOW + b2 + e] : 0.0f;
                Tf[t][cbi][e]     = (__bf16)p1;
                Tf[t][cbi][e + 8] = (__bf16)p2;
            }
        }
    }

    // ---- per-lane bin biases (24 registers), -inf on padded bins ----
    float bias0[8], bias1[8], bias2[8];
    #pragma unroll
    for (int r = 0; r < 8; ++r) {
        const int kb = 8 * h + r;
        bias0[r] = sBias[kb];
        bias1[r] = sBias[16 + kb];
        bias2[r] = (32 + kb < NUM_BINS) ? sBias[32 + kb] : NEG_PAD;
    }

    const float xj0 = xtrue[(size_t)(bN + j) * 3 + 0];
    const float xj1 = xtrue[(size_t)(bN + j) * 3 + 1];
    const float xj2 = xtrue[(size_t)(bN + j) * 3 + 2];
    const float mj  = pmask[bN + j];

    float thCe = 0.0f, thCnt = 0.0f;

    const int it0 = blockIdx.y * itilesPerSplit;
    for (int it = it0; it < it0 + itilesPerSplit; ++it) {
        const int i0 = it * 16;

        // ---- B fragment: U columns i0..i0+15, 64-deep bf16 ----
        const unsigned short* up = Ubf + (size_t)(bN + i0 + col) * D_LOW;
        v16bf Uf[2];
        #pragma unroll
        for (int cbi = 0; cbi < 2; ++cbi) {
            const int b1 = cbi * 32 + h * 8;
            const int b2 = b1 + 16;
            #pragma unroll
            for (int e = 0; e < 8; ++e) {
                Uf[cbi][e]     = bfbits(up[b1 + e]);
                Uf[cbi][e + 8] = bfbits(up[b2 + e]);
            }
        }

        // ---- 6 WMMAs: C[k][i] = T[j] x U, 48 x 16, f32 accumulate ----
        v8f a0 = {}, a1 = {}, a2 = {};
        a0 = __builtin_amdgcn_wmma_f32_16x16x32_bf16(false, Tf[0][0], false, Uf[0], (short)0, a0, false, false);
        a0 = __builtin_amdgcn_wmma_f32_16x16x32_bf16(false, Tf[0][1], false, Uf[1], (short)0, a0, false, false);
        a1 = __builtin_amdgcn_wmma_f32_16x16x32_bf16(false, Tf[1][0], false, Uf[0], (short)0, a1, false, false);
        a1 = __builtin_amdgcn_wmma_f32_16x16x32_bf16(false, Tf[1][1], false, Uf[1], (short)0, a1, false, false);
        a2 = __builtin_amdgcn_wmma_f32_16x16x32_bf16(false, Tf[2][0], false, Uf[0], (short)0, a2, false, false);
        a2 = __builtin_amdgcn_wmma_f32_16x16x32_bf16(false, Tf[2][1], false, Uf[1], (short)0, a2, false, false);

        // ---- per-lane pair (i = i0+col, j): 24 bins in regs, 24 in lane^16 ----
        const int i = i0 + col;
        const float mi = pmask[bN + i];
        const float dx = xtrue[(size_t)(bN + i) * 3 + 0] - xj0;
        const float dy = xtrue[(size_t)(bN + i) * 3 + 1] - xj1;
        const float dz = xtrue[(size_t)(bN + i) * 3 + 2] - xj2;
        const float d  = sqrtf(dx * dx + dy * dy + dz * dz);
        int tk = (int)((d - DIST_MIN) * INV_BINW);     // trunc like .astype(int32)
        tk = tk < 0 ? 0 : (tk > NUM_BINS - 1 ? NUM_BINS - 1 : tk);
        const int tkr = tk - 8 * h;                     // lane-relative target row

        float l0[8], l1[8], l2[8];
        float mloc = NEG_PAD;
        #pragma unroll
        for (int r = 0; r < 8; ++r) {
            l0[r] = a0[r] + bias0[r];
            l1[r] = a1[r] + bias1[r];
            l2[r] = a2[r] + bias2[r];
            mloc = fmaxf(mloc, fmaxf(fmaxf(l0[r], l1[r]), l2[r]));
        }
        const float mx = fmaxf(mloc, xhalf(mloc));

        float sloc = 0.0f, selloc = 0.0f;
        #pragma unroll
        for (int r = 0; r < 8; ++r) {
            sloc += __expf(l0[r] - mx);
            sloc += __expf(l1[r] - mx);
            sloc += __expf(l2[r] - mx);
            selloc += (r      == tkr) ? l0[r] : 0.0f;
            selloc += (r + 16 == tkr) ? l1[r] : 0.0f;
            selloc += (r + 32 == tkr) ? l2[r] : 0.0f;
        }
        const float s   = sloc   + xhalf(sloc);
        const float sel = selloc + xhalf(selloc);

        if (h == 0 && (mi * mj > 0.0f)) {
            thCe  += mx + __logf(s) - sel;   // -(logit_tk - mx - log S)
            thCnt += 1.0f;
        }
    }

    sCe[tid] = thCe; sCnt[tid] = thCnt;
    __syncthreads();
    #pragma unroll
    for (int off = 128; off > 0; off >>= 1) {
        if (tid < off) { sCe[tid] += sCe[tid + off]; sCnt[tid] += sCnt[tid + off]; }
        __syncthreads();
    }
    if (tid == 0) {
        atomicAdd(&acc4[bb * 2 + 0], sCe[0]);
        atomicAdd(&acc4[bb * 2 + 1], sCnt[0]);
    }
}

// ---------------------------------------------------------------------------
// Kernel 3: finalize scalar loss (mean over valid batches)
// ---------------------------------------------------------------------------
__global__ void k_final(const float* __restrict__ acc4, float* __restrict__ out) {
    float s = 0.0f; int valid = 0;
    #pragma unroll
    for (int b = 0; b < 2; ++b) {
        const float c = acc4[2 * b + 1];
        if (c > 0.0f) { s += acc4[2 * b] / c; ++valid; }
    }
    out[0] = (valid > 0) ? s / (float)valid : 0.0f;
}

// ---------------------------------------------------------------------------
extern "C" void kernel_launch(void* const* d_in, const int* in_sizes, int n_in,
                              void* d_out, int out_size, void* d_ws, size_t ws_size,
                              hipStream_t stream)
{
    const float* hres  = (const float*)d_in[0];
    const float* xtrue = (const float*)d_in[1];
    const float* pm    = (const float*)d_in[2];
    const float* lnw   = (const float*)d_in[3];
    const float* lnb   = (const float*)d_in[4];
    const float* wuw   = (const float*)d_in[5];
    const float* wub   = (const float*)d_in[6];
    const float* wvw   = (const float*)d_in[7];
    const float* wvb   = (const float*)d_in[8];
    const float* wbw   = (const float*)d_in[9];
    const float* wbb   = (const float*)d_in[10];

    const int BN = in_sizes[2];   // token_pad_mask has B*N elements
    const int B  = 2;             // fixed by reference
    const int N  = BN / B;

    // workspace layout: [0,16): batch accumulators; [256,..): U bf16; then V f32
    float*          acc4 = (float*)d_ws;
    unsigned short* Ubf  = (unsigned short*)((char*)d_ws + 256);
    float*          Vf   = (float*)((char*)d_ws + 256 + (size_t)BN * D_LOW * sizeof(unsigned short));

    k_init<<<1, 32, 0, stream>>>(acc4);
    k_lnproj<<<BN, 256, 0, stream>>>(hres, lnw, lnb, wuw, wub, wvw, wvb, Ubf, Vf);

    const int jgPerB = N / 8;     // 8 waves per block, one j per wave
    const int SPLITS = 4;         // split i-range for occupancy
    const int itiles = N / 16;
    dim3 grid(B * jgPerB, SPLITS);
    k_loss<<<grid, 256, 0, stream>>>(Ubf, Vf, xtrue, pm, wbw, wbb,
                                     acc4, N, jgPerB, itiles / SPLITS);

    k_final<<<1, 1, 0, stream>>>(acc4, (float*)d_out);
}
